// CSDI_GRIN_44633300140827
// MI455X (gfx1250) — matrix operand order repaired and checked by reading, not compiled, not run on hardware
//
#include <hip/hip_runtime.h>
#include <hip/hip_bf16.h>
#include <math.h>

typedef __attribute__((ext_vector_type(16))) _Float16 v16h;
typedef __attribute__((ext_vector_type(8)))  float    v8f;
typedef _Float16 h16;

// ---------------- problem constants ----------------
constexpr int Hh   = 64;     // hidden
constexpr int EMBn = 128;    // diffusion embedding
constexpr int Kn   = 207;    // nodes
constexpr int Bn   = 8;      // batch
constexpr int Ln   = 48;     // sequence length
constexpr int KP   = 256;    // padded nodes (multiple of 64)
constexpr int NBc  = Bn * KP;            // 2048 columns for step GEMMs
constexpr long NCOL = (long)NBc * Ln;    // 98304 columns for final GEMM
constexpr int GINc = 5 * KP;             // 1280 padded gate input channels
constexpr int RFP  = 320;                // padded repr channels (4H+2 -> 320)

// ---------------- WMMA GEMM: out[M,N] = act(A[M,K]*B[K,N] + bias) ----------------
// 64x64 block tile, 4 waves x 4 accumulators, double-buffered LDS filled with
// gfx1250 async direct-to-LDS loads (ASYNCcnt), fragments via ds_load_b128.
// ACT: 0 none, 1 sigmoid, 2 tanh, 3 relu, 4 prelu(actp[0]).
template<int ACT, bool HASBIAS, bool F16OUT>
__global__ __launch_bounds__(128)
void k_gemm(const h16* __restrict__ A, int lda, long aBatch,
            const h16* __restrict__ Bm, int ldb, long bBatch,
            void* __restrict__ outp, int ldo, long oBatch,
            int Kin, const float* __restrict__ bias, const float* __restrict__ actp)
{
    __shared__ h16 sa[2][64 * 32];
    __shared__ h16 sb[2][32 * 64];
    const int bb   = blockIdx.z;
    const h16* Ap  = A  + (long)bb * aBatch;
    const h16* Bp  = Bm + (long)bb * bBatch;
    const int m0   = blockIdx.y * 64;
    const int n0   = blockIdx.x * 64;
    const int tid  = threadIdx.x;
    const int lane = tid & 31;
    const int wid  = tid >> 5;
    const int sy   = (wid & 1) * 16;
    const int sx   = (wid >> 1) * 16;
    v8f acc00 = {}, acc01 = {}, acc10 = {}, acc11 = {};
    const int nk = Kin >> 5;

    // per-thread staging slots: 2 x 16B of A-tile (64x32) and 2 x 16B of B-tile (32x64)
    const int ar0 = tid >> 2,          ac0 = (tid & 3) * 8;
    const int ar1 = (tid + 128) >> 2,  ac1 = ((tid + 128) & 3) * 8;
    const int br0 = tid >> 3,          bc0 = (tid & 7) * 8;
    const int br1 = (tid + 128) >> 3,  bc1 = ((tid + 128) & 7) * 8;
    const h16* gA0 = Ap + (long)(m0 + ar0) * lda + ac0;
    const h16* gA1 = Ap + (long)(m0 + ar1) * lda + ac1;
    const h16* gB0 = Bp + (long)br0 * ldb + n0 + bc0;
    const h16* gB1 = Bp + (long)br1 * ldb + n0 + bc1;
    // LDS byte offsets (generic shared address truncates to LDS offset)
    const unsigned lA0 = (unsigned)(size_t)&sa[0][ar0 * 32 + ac0];
    const unsigned lA1 = (unsigned)(size_t)&sa[0][ar1 * 32 + ac1];
    const unsigned lB0 = (unsigned)(size_t)&sb[0][br0 * 64 + bc0];
    const unsigned lB1 = (unsigned)(size_t)&sb[0][br1 * 64 + bc1];

    auto stage = [&](int ks, int buf) {
        const long kc = (long)ks << 5;
        const unsigned o = (unsigned)buf * 4096u;   // 2048 halves per buffer
        const h16* pa0 = gA0 + kc;
        const h16* pa1 = gA1 + kc;
        const h16* pb0 = gB0 + kc * ldb;
        const h16* pb1 = gB1 + kc * ldb;
        asm volatile("global_load_async_to_lds_b128 %0, %1, off"
                     :: "v"(lA0 + o), "v"(pa0) : "memory");
        asm volatile("global_load_async_to_lds_b128 %0, %1, off"
                     :: "v"(lA1 + o), "v"(pa1) : "memory");
        asm volatile("global_load_async_to_lds_b128 %0, %1, off"
                     :: "v"(lB0 + o), "v"(pb0) : "memory");
        asm volatile("global_load_async_to_lds_b128 %0, %1, off"
                     :: "v"(lB1 + o), "v"(pb1) : "memory");
        if (ks + 1 < nk) {                          // prefetch one stage further ahead
            __builtin_prefetch(pa0 + 32, 0, 3);
            __builtin_prefetch(pb0 + (long)32 * ldb, 0, 3);
        }
    };

    stage(0, 0);
    asm volatile("s_wait_asynccnt 0x0" ::: "memory");
    __syncthreads();
    for (int ks = 0; ks < nk; ++ks) {
        const int buf = ks & 1;
        const int am   = lane & 15;
        const int koff = (lane & 16) ? 8 : 0;
        v16h a0, a1, b0, b1;
#pragma unroll
        for (int j = 0; j < 8; ++j) {                          // CDNA5 16-bit A 16x32 layout
            int k = ((j >= 4) ? 16 : 0) + koff + 2 * (j & 3);
            a0[2 * j]     = sa[buf][(sy + am) * 32 + k];
            a0[2 * j + 1] = sa[buf][(sy + am) * 32 + k + 1];
            a1[2 * j]     = sa[buf][(sy + 32 + am) * 32 + k];
            a1[2 * j + 1] = sa[buf][(sy + 32 + am) * 32 + k + 1];
        }
#pragma unroll
        for (int e = 0; e < 16; ++e) {                         // B: lane = N column, K in table order
            int kb = ((e >= 8) ? 16 : 0) + koff + (e & 7);
            b0[e] = sb[buf][kb * 64 + sx + am];
            b1[e] = sb[buf][kb * 64 + sx + 32 + am];
        }
        acc00 = __builtin_amdgcn_wmma_f32_16x16x32_f16(false, a0, false, b0, (short)0, acc00, false, false);
        acc01 = __builtin_amdgcn_wmma_f32_16x16x32_f16(false, a0, false, b1, (short)0, acc01, false, false);
        acc10 = __builtin_amdgcn_wmma_f32_16x16x32_f16(false, a1, false, b0, (short)0, acc10, false, false);
        acc11 = __builtin_amdgcn_wmma_f32_16x16x32_f16(false, a1, false, b1, (short)0, acc11, false, false);
        if (ks + 1 < nk) stage(ks + 1, buf ^ 1);
        asm volatile("s_wait_asynccnt 0x0" ::: "memory");      // drain our async LDS writes
        __syncthreads();
    }

    const int ccol  = lane & 15;
    const int rbase = (lane & 16) ? 8 : 0;
    auto emit = [&](v8f& a, int ry, int cx) {
#pragma unroll
        for (int r = 0; r < 8; ++r) {
            int row = m0 + ry + rbase + r;
            int col = n0 + cx + ccol;
            float v = a[r];
            if (HASBIAS) v += bias[row];
            if (ACT == 1)      v = 1.f / (1.f + __expf(-v));
            else if (ACT == 2) v = tanhf(v);
            else if (ACT == 3) v = fmaxf(v, 0.f);
            else if (ACT == 4) { float al = actp[0]; v = v > 0.f ? v : al * v; }
            long o = (long)row * ldo + col + (long)bb * oBatch;
            if (F16OUT) ((h16*)outp)[o] = (h16)v;
            else        ((float*)outp)[o] = v;
        }
    };
    emit(acc00, sy, sx);
    emit(acc01, sy, sx + 32);
    emit(acc10, sy + 32, sx);
    emit(acc11, sy + 32, sx + 32);
}

// ---------------- diffusion embedding MLP: e[B][128] ----------------
__global__ __launch_bounds__(128)
void k_emb(const int* __restrict__ step,
           const float* __restrict__ W1, const float* __restrict__ b1,
           const float* __restrict__ W2, const float* __restrict__ b2,
           float* __restrict__ e)
{
    const int b = blockIdx.x, t = threadIdx.x;
    __shared__ float s0[EMBn], s1[EMBn];
    const int s = step[b];
    const int halfd = EMBn / 2;
    float val;
    if (t < halfd) {
        float fr = __powf(10.f, (float)t / (float)(halfd - 1) * 4.f);
        val = __sinf((float)s * fr);
    } else {
        float fr = __powf(10.f, (float)(t - halfd) / (float)(halfd - 1) * 4.f);
        val = __cosf((float)s * fr);
    }
    s0[t] = val; __syncthreads();
    float a = b1[t];
    for (int j = 0; j < EMBn; ++j) a += W1[t * EMBn + j] * s0[j];
    a = a / (1.f + __expf(-a));
    s1[t] = a; __syncthreads();
    float c = b2[t];
    for (int j = 0; j < EMBn; ++j) c += W2[t * EMBn + j] * s1[j];
    c = c / (1.f + __expf(-c));
    e[b * EMBn + t] = c;
}

// ---------------- supports (A symmetric -> a_fwd == a_bwd) ----------------
__global__ void k_rowsum(float* __restrict__ rs)
{
    int w = blockIdx.x * blockDim.x + threadIdx.x;
    if (w >= Kn) return;
    float s = 0.f;
    for (int v = 0; v < Kn; ++v) {
        float d = (float)(w - v);
        float a = __expf(-0.5f * d * d);
        if (v == w) a -= 1.f;
        s += a;
    }
    rs[w] = s + 1e-8f;
}
__global__ void k_fill_aT(const float* __restrict__ rs, h16* __restrict__ aT)
{
    long i = (long)blockIdx.x * blockDim.x + threadIdx.x;
    if (i >= (long)KP * KP) return;
    int v = (int)(i / KP), w = (int)(i % KP);
    float val = 0.f;
    if (v < Kn && w < Kn) {
        float d = (float)(w - v);
        float a = __expf(-0.5f * d * d);
        if (v == w) a -= 1.f;
        val = a / rs[w];
    }
    aT[i] = (h16)val;   // aT[v][w] = a_norm[w][v]
}

// ---------------- weight prep ----------------
__global__ void k_padw(const float* __restrict__ src, h16* __restrict__ dst,
                       int M, int sc, int dc, int rowOff)
{
    long i = (long)blockIdx.x * blockDim.x + threadIdx.x;
    if (i >= (long)M * dc) return;
    int r = (int)(i / dc), c = (int)(i % dc);
    dst[(long)(rowOff + r) * dc + c] = (c < sc) ? (h16)src[(long)r * sc + c] : (h16)0.f;
}
__global__ void k_padgate(const float* __restrict__ src, h16* __restrict__ dst,
                          int M, int rowOff)
{
    long i = (long)blockIdx.x * blockDim.x + threadIdx.x;
    if (i >= (long)M * GINc) return;
    int r = (int)(i / GINc), c = (int)(i % GINc);
    int blk = c / KP, j = c % KP;
    float v = (j < 211) ? src[(long)r * 1055 + blk * 211 + j] : 0.f;
    dst[(long)(rowOff + r) * GINc + c] = (h16)v;
}
__global__ void k_concat2(const float* a, const float* b, float* dst)
{
    int t = threadIdx.x;
    if (t < 64) dst[t] = a[t];
    else if (t < 128) dst[t] = b[t - 64];
}
__global__ void k_padvecf(const float* src, float* dst, int n, int np)
{
    int t = blockIdx.x * blockDim.x + threadIdx.x;
    if (t < np) dst[t] = (t < n) ? src[t] : 0.f;
}
__global__ void k_zerof(float* p, long n)
{
    long i = (long)blockIdx.x * blockDim.x + threadIdx.x;
    if (i < n) p[i] = 0.f;
}

// ---------------- per-step glue ----------------
// build dec_in = [x1, m, h, u, v] (256 rows padded), x1 = m ? x : W_fs.h + b_fs
__global__ __launch_bounds__(64)
void k_step_pre(const float* __restrict__ hbuf, const float* __restrict__ e,
                const float* __restrict__ cond_obs, const float* __restrict__ cond_mask,
                const float* __restrict__ side, const float* __restrict__ noisy,
                const float* __restrict__ W_fs, const float* __restrict__ b_fs,
                h16* __restrict__ dec, int treal)
{
    int colk = blockIdx.x;
    int b = colk / Kn, k = colk % Kn;
    int col = b * KP + k;
    int tid = threadIdx.x;
    __shared__ float red[64], sh[64];
    float hv = hbuf[(long)tid * NBc + col];
    sh[tid] = hv;
    red[tid] = W_fs[tid] * hv;
    __syncthreads();
    for (int s = 32; s > 0; s >>= 1) { if (tid < s) red[tid] += red[tid + s]; __syncthreads(); }
    float xh1 = red[0] + b_fs[0];
    long tix = ((long)b * Kn + k) * Ln + treal;
    float x = cond_obs[tix], m = cond_mask[tix], vn = noisy[tix];
    float x1 = (m > 0.5f) ? x : xh1;
    for (int r = tid; r < KP; r += 64) {
        float val;
        if (r == 0) val = x1;
        else if (r == 1) val = m;
        else if (r < 66) val = sh[r - 2];
        else if (r < 210) {
            int c = r - 66;
            val = (c < 16) ? side[((long)(b * 16 + c) * Kn + k) * Ln + treal]
                           : e[b * EMBn + (c - 16)];
        }
        else if (r == 210) val = vn;
        else val = 0.f;
        dec[(long)r * NBc + col] = (h16)val;
    }
}

// xh2 = W_ro.[z,h]+b; x2 = m ? x : xh2; build gate block0 = [x2,m,u,v,h]; stash repr=[z,h]
__global__ __launch_bounds__(64)
void k_dec_out(const h16* __restrict__ z, const float* __restrict__ hbuf,
               const float* __restrict__ e,
               const float* __restrict__ cond_obs, const float* __restrict__ cond_mask,
               const float* __restrict__ side, const float* __restrict__ noisy,
               const float* __restrict__ W_ro, const float* __restrict__ b_ro,
               h16* __restrict__ g2in, h16* __restrict__ repr, int dirbase, int treal)
{
    int colk = blockIdx.x;
    int b = colk / Kn, k = colk % Kn;
    int col = b * KP + k;
    int tid = threadIdx.x;
    __shared__ float red[64], sh[64];
    float zv = (float)z[(long)tid * NBc + col];
    float hv = hbuf[(long)tid * NBc + col];
    sh[tid] = hv;
    red[tid] = W_ro[tid] * zv + W_ro[64 + tid] * hv;
    __syncthreads();
    for (int s = 32; s > 0; s >>= 1) { if (tid < s) red[tid] += red[tid + s]; __syncthreads(); }
    float xh2 = red[0] + b_ro[0];
    long tix = ((long)b * Kn + k) * Ln + treal;
    float x = cond_obs[tix], m = cond_mask[tix], vn = noisy[tix];
    float x2 = (m > 0.5f) ? x : xh2;
    for (int r = tid; r < KP; r += 64) {
        float val;
        if (r == 0) val = x2;
        else if (r == 1) val = m;
        else if (r < 146) {
            int c = r - 2;
            val = (c < 16) ? side[((long)(b * 16 + c) * Kn + k) * Ln + treal]
                           : e[b * EMBn + (c - 16)];
        }
        else if (r == 146) val = vn;
        else if (r < 211) val = sh[r - 147];
        else val = 0.f;
        g2in[(long)r * NBc + col] = (h16)val;
    }
    long colr = (long)col * Ln + treal;
    repr[(long)(dirbase + tid) * NCOL + colr]      = (h16)zv;
    repr[(long)(dirbase + 64 + tid) * NCOL + colr] = (h16)hv;
}

__global__ __launch_bounds__(256)
void k_make_rh(const float* __restrict__ rg, const float* __restrict__ hbuf,
               h16* __restrict__ rh, h16* __restrict__ g2in)
{
    long i = (long)blockIdx.x * blockDim.x + threadIdx.x;
    if (i >= (long)Hh * NBc) return;
    int r = (int)(i / NBc), col = (int)(i % NBc);
    float val = rg[(long)r * NBc + col] * hbuf[(long)r * NBc + col];
    rh[i] = (h16)val;
    g2in[(long)(147 + r) * NBc + col] = (h16)val;
}

__global__ __launch_bounds__(256)
void k_update_h(const float* __restrict__ rg, const float* __restrict__ cbuf,
                float* __restrict__ hbuf)
{
    long i = (long)blockIdx.x * blockDim.x + threadIdx.x;
    if (i >= (long)Hh * NBc) return;
    int r = (int)(i / NBc), col = (int)(i % NBc);
    float g = rg[(long)(64 + r) * NBc + col];
    hbuf[i] = g * hbuf[i] + (1.f - g) * cbuf[i];
}

// ---------------- final merge ----------------
__global__ __launch_bounds__(256)
void k_fill_mv(const float* __restrict__ cond_mask, const float* __restrict__ noisy,
               h16* __restrict__ repr)
{
    long col = (long)blockIdx.x * blockDim.x + threadIdx.x;
    if (col >= NCOL) return;
    int l = (int)(col % Ln);
    long bk = col / Ln;
    int k = (int)(bk % KP), b = (int)(bk / KP);
    float m = 0.f, vn = 0.f;
    if (k < Kn) {
        long tix = ((long)b * Kn + k) * Ln + l;
        m = cond_mask[tix]; vn = noisy[tix];
    }
    repr[(long)256 * NCOL + col] = (h16)m;
    repr[(long)257 * NCOL + col] = (h16)vn;
    for (int r = 258; r < RFP; ++r) repr[(long)r * NCOL + col] = (h16)0.f;
}

__global__ __launch_bounds__(256)
void k_final(const h16* __restrict__ hidden, const h16* __restrict__ repr,
             const float* __restrict__ W2, const float* __restrict__ b2,
             const float* __restrict__ outWp, const float* __restrict__ outb,
             float* __restrict__ out)
{
    long i = (long)blockIdx.x * blockDim.x + threadIdx.x;
    const long TOT = (long)Bn * Kn * Ln;
    if (i >= TOT) return;
    int l = (int)(i % Ln);
    long bk = i / Ln;
    int k = (int)(bk % Kn), b = (int)(bk / Kn);
    long col = (long)(b * KP + k) * Ln + l;
    float acc1 = b2[0];
    for (int j = 0; j < 256; ++j) acc1 += W2[j] * (float)hidden[(long)j * NCOL + col];
    float acc2 = outb[0];
    for (int j = 0; j < RFP; ++j) acc2 += outWp[j] * (float)repr[(long)j * NCOL + col];
    out[i]       = acc2;   // y (noise prediction)
    out[TOT + i] = acc1;   // imputation
}

// =======================================================================
extern "C" void kernel_launch(void* const* d_in, const int* in_sizes, int n_in,
                              void* d_out, int out_size, void* d_ws, size_t ws_size,
                              hipStream_t stream)
{
    (void)in_sizes; (void)n_in; (void)out_size; (void)ws_size;
    const float* cond_obs  = (const float*)d_in[0];
    const float* cond_mask = (const float*)d_in[1];
    const float* side      = (const float*)d_in[2];
    const float* noisy     = (const float*)d_in[3];
    const int*   dstep     = (const int*)d_in[4];
    const float* emb_W1 = (const float*)d_in[5];
    const float* emb_b1 = (const float*)d_in[6];
    const float* emb_W2 = (const float*)d_in[7];
    const float* emb_b2 = (const float*)d_in[8];
    auto P = [&](int dir, int j) { return (const float*)d_in[9 + dir * 15 + j]; };
    // j: 0 Wr 1 br 2 Wu 3 bu 4 Wc 5 bc 6 W_fs 7 b_fs 8 W_lin 9 b_lin 10 W_gc 11 b_gc 12 W_ro 13 b_ro 14 prelu
    const float* mlp_W1 = (const float*)d_in[39];
    const float* mlp_b1 = (const float*)d_in[40];
    const float* mlp_W2 = (const float*)d_in[41];
    const float* mlp_b2 = (const float*)d_in[42];
    const float* out_W  = (const float*)d_in[43];
    const float* out_b  = (const float*)d_in[44];

    // -------- workspace layout --------
    char* w = (char*)d_ws;
    size_t off = 0;
    auto Al = [&](size_t bytes) { void* p = w + off; off += (bytes + 255) & ~(size_t)255; return p; };
    h16*   aT      = (h16*)  Al((size_t)KP * KP * 2);
    float* ebuf    = (float*)Al((size_t)Bn * EMBn * 4);
    float* rowsum  = (float*)Al((size_t)Kn * 4);
    h16* Wlin_h[2]; h16* Wgc_h[2]; h16* Wru_h[2]; h16* Wc_h[2]; float* brbu[2];
    for (int d = 0; d < 2; ++d) {
        Wlin_h[d] = (h16*)  Al((size_t)64 * KP * 2);
        Wgc_h[d]  = (h16*)  Al((size_t)64 * 128 * 2);
        Wru_h[d]  = (h16*)  Al((size_t)128 * GINc * 2);
        Wc_h[d]   = (h16*)  Al((size_t)64 * GINc * 2);
        brbu[d]   = (float*)Al((size_t)128 * 4);
    }
    h16*   mlpW1_h = (h16*)  Al((size_t)256 * RFP * 2);
    float* outWp   = (float*)Al((size_t)RFP * 4);
    h16*   dec     = (h16*)  Al((size_t)KP * NBc * 2);
    h16*   z0      = (h16*)  Al((size_t)64 * NBc * 2);
    h16*   zc      = (h16*)  Al((size_t)128 * NBc * 2);
    h16*   zbuf    = (h16*)  Al((size_t)64 * NBc * 2);
    h16*   g2in    = (h16*)  Al((size_t)GINc * NBc * 2);
    h16*   rh      = (h16*)  Al((size_t)64 * NBc * 2);
    float* rg      = (float*)Al((size_t)128 * NBc * 4);
    float* cbuf    = (float*)Al((size_t)64 * NBc * 4);
    float* hbuf    = (float*)Al((size_t)64 * NBc * 4);
    h16*   repr    = (h16*)  Al((size_t)RFP * NCOL * 2);
    h16*   hidden  = (h16*)  Al((size_t)256 * NCOL * 2);

    // templated GEMM dispatch
    auto gemm = [&](const h16* Am, int lda, long aB, const h16* Bm, int ldb, long bB,
                    void* outp, bool f16out, int ldo, long oB,
                    int M, long N, int Kin, int nb,
                    const float* bias, int act, const float* actp) {
        dim3 g((unsigned)(N / 64), (unsigned)(M / 64), (unsigned)nb);
#define LK(ACT, HB, F16) k_gemm<ACT, HB, F16><<<g, 128, 0, stream>>>( \
        Am, lda, aB, Bm, ldb, bB, outp, ldo, oB, Kin, bias, actp)
        if (f16out) {
            if (!bias)            LK(0, false, true);   // spatial
            else if (act == 0)    LK(0, true,  true);   // W_lin
            else if (act == 3)    LK(3, true,  true);   // merge MLP relu
            else                  LK(4, true,  true);   // W_gc prelu
        } else {
            if (act == 1)         LK(1, true,  false);  // gates sigmoid
            else                  LK(2, true,  false);  // candidate tanh
        }
#undef LK
    };

    // -------- prep --------
    k_emb<<<Bn, EMBn, 0, stream>>>(dstep, emb_W1, emb_b1, emb_W2, emb_b2, ebuf);
    k_rowsum<<<1, 256, 0, stream>>>(rowsum);
    k_fill_aT<<<(KP * KP + 255) / 256, 256, 0, stream>>>(rowsum, aT);
    for (int d = 0; d < 2; ++d) {
        int gg = (64 * GINc + 255) / 256;
        k_padgate<<<gg, 256, 0, stream>>>(P(d, 0), Wru_h[d], 64, 0);
        k_padgate<<<gg, 256, 0, stream>>>(P(d, 2), Wru_h[d], 64, 64);
        k_padgate<<<gg, 256, 0, stream>>>(P(d, 4), Wc_h[d], 64, 0);
        k_padw<<<(64 * KP + 255) / 256, 256, 0, stream>>>(P(d, 8), Wlin_h[d], 64, 211, KP, 0);
        k_padw<<<(64 * 128 + 255) / 256, 256, 0, stream>>>(P(d, 10), Wgc_h[d], 64, 128, 128, 0);
        k_concat2<<<1, 128, 0, stream>>>(P(d, 1), P(d, 3), brbu[d]);
    }
    k_padw<<<(256 * RFP + 255) / 256, 256, 0, stream>>>(mlp_W1, mlpW1_h, 256, 258, RFP, 0);
    k_padvecf<<<1, RFP, 0, stream>>>(out_W, outWp, 258, RFP);

    // -------- bidirectional GRIL --------
    for (int dir = 0; dir < 2; ++dir) {
        k_zerof<<<((long)64 * NBc + 255) / 256, 256, 0, stream>>>(hbuf, (long)64 * NBc);
        const float* Wfs = P(dir, 6),  *bfs = P(dir, 7);
        const float* Wro = P(dir, 12), *bro = P(dir, 13);
        const float* prelu = P(dir, 14);
        int dirbase = dir ? 128 : 0;
        for (int t = 0; t < Ln; ++t) {
            int treal = dir ? (Ln - 1 - t) : t;
            k_step_pre<<<Bn * Kn, 64, 0, stream>>>(hbuf, ebuf, cond_obs, cond_mask, side,
                                                   noisy, Wfs, bfs, dec, treal);
            // z0 = W_lin * dec_in + b_lin
            gemm(Wlin_h[dir], KP, 0, dec, NBc, 0, z0, true, NBc, 0,
                 64, NBc, KP, 1, P(dir, 9), 0, nullptr);
            // gconv1 spatial: zc = [z0@a1T ; z0@a2T]  (a1==a2 by symmetry)
            gemm(z0, NBc, KP, aT, KP, 0, zc, true, NBc, KP, 64, KP, KP, Bn, nullptr, 0, nullptr);
            gemm(z0, NBc, KP, aT, KP, 0, zc + (size_t)64 * NBc, true, NBc, KP,
                 64, KP, KP, Bn, nullptr, 0, nullptr);
            // z = prelu(W_gc * zc + b_gc)
            gemm(Wgc_h[dir], 128, 0, zc, NBc, 0, zbuf, true, NBc, 0,
                 64, NBc, 128, 1, P(dir, 11), 4, prelu);
            k_dec_out<<<Bn * Kn, 64, 0, stream>>>(zbuf, hbuf, ebuf, cond_obs, cond_mask, side,
                                                  noisy, Wro, bro, g2in, repr, dirbase, treal);
            // gconv2 spatial chain on xh=[cin,h]
            gemm(g2in, NBc, KP, aT, KP, 0, g2in + (size_t)KP * NBc,     true, NBc, KP, KP, KP, KP, Bn, nullptr, 0, nullptr);
            gemm(g2in, NBc, KP, aT, KP, 0, g2in + (size_t)3 * KP * NBc, true, NBc, KP, KP, KP, KP, Bn, nullptr, 0, nullptr);
            gemm(g2in + (size_t)KP * NBc,     NBc, KP, aT, KP, 0, g2in + (size_t)2 * KP * NBc, true, NBc, KP, KP, KP, KP, Bn, nullptr, 0, nullptr);
            gemm(g2in + (size_t)3 * KP * NBc, NBc, KP, aT, KP, 0, g2in + (size_t)4 * KP * NBc, true, NBc, KP, KP, KP, KP, Bn, nullptr, 0, nullptr);
            // r,g = sigmoid([Wr;Wu] * g2in + [br;bu])
            gemm(Wru_h[dir], GINc, 0, g2in, NBc, 0, rg, false, NBc, 0,
                 128, NBc, GINc, 1, brbu[dir], 1, nullptr);
            // overwrite h rows of every block with r*h versions
            k_make_rh<<<((long)64 * NBc + 255) / 256, 256, 0, stream>>>(rg, hbuf, rh, g2in);
            gemm(rh, NBc, KP, aT, KP, 0, g2in + ((size_t)KP + 147) * NBc,     true, NBc, KP, 64, KP, KP, Bn, nullptr, 0, nullptr);
            gemm(rh, NBc, KP, aT, KP, 0, g2in + ((size_t)3 * KP + 147) * NBc, true, NBc, KP, 64, KP, KP, Bn, nullptr, 0, nullptr);
            gemm(g2in + ((size_t)KP + 147) * NBc,     NBc, KP, aT, KP, 0, g2in + ((size_t)2 * KP + 147) * NBc, true, NBc, KP, 64, KP, KP, Bn, nullptr, 0, nullptr);
            gemm(g2in + ((size_t)3 * KP + 147) * NBc, NBc, KP, aT, KP, 0, g2in + ((size_t)4 * KP + 147) * NBc, true, NBc, KP, 64, KP, KP, Bn, nullptr, 0, nullptr);
            // c = tanh(Wc * g2in + bc)
            gemm(Wc_h[dir], GINc, 0, g2in, NBc, 0, cbuf, false, NBc, 0,
                 64, NBc, GINc, 1, P(dir, 5), 2, nullptr);
            k_update_h<<<((long)64 * NBc + 255) / 256, 256, 0, stream>>>(rg, cbuf, hbuf);
        }
    }

    // -------- merge MLP + outputs --------
    k_fill_mv<<<(unsigned)((NCOL + 255) / 256), 256, 0, stream>>>(cond_mask, noisy, repr);
    gemm(mlpW1_h, RFP, 0, repr, (int)NCOL, 0, hidden, true, (int)NCOL, 0,
         256, NCOL, RFP, 1, mlp_b1, 3, nullptr);
    k_final<<<(Bn * Kn * Ln + 255) / 256, 256, 0, stream>>>(hidden, repr, mlp_W2, mlp_b2,
                                                            outWp, out_b, (float*)d_out);
}
